// KANLinear_9302899163590
// MI455X (gfx1250) — matrix operand order repaired and checked by reading, not compiled, not run on hardware
//
#include <hip/hip_runtime.h>

typedef __bf16 bf16_t;
typedef __attribute__((ext_vector_type(8)))  __bf16       v8bf;
typedef __attribute__((ext_vector_type(16))) __bf16       v16bf;
typedef __attribute__((ext_vector_type(8)))  float        v8f;
typedef __attribute__((ext_vector_type(4)))  unsigned int v4u;
typedef int v4i_t __attribute__((vector_size(4 * sizeof(int))));

#define IN_F   1024
#define OUT_F  1024
#define NB     8
#define KTOT   (IN_F + IN_F * NB)   /* 9216 */
#define BATCH  8192

#define BM 128
#define BN 256
#define BK 32
#define LDSK 40   /* padded LDS row stride in bf16 (80 B) -> conflict-free b128 loads */
#define NBUF 3    /* 3-stage async pipeline, issue distance 2 */

#define ASG __attribute__((address_space(1)))
#define ASL __attribute__((address_space(3)))

#if defined(__has_builtin)
#if __has_builtin(__builtin_amdgcn_global_load_async_to_lds_b128)
#define HAS_ASYNC_LDS 1
#endif
#endif
#ifndef HAS_ASYNC_LDS
#define HAS_ASYNC_LDS 0
#endif

// copy 16 B from g+OFF to l+OFF (async DMA to LDS; INST_OFFSET is added to
// BOTH the global and LDS addresses by the hardware)
template <int OFF>
__device__ __forceinline__ void cp16(const bf16_t* g, bf16_t* l)
{
#if HAS_ASYNC_LDS
    __builtin_amdgcn_global_load_async_to_lds_b128(
        (ASG v4i_t*)(const void*)g, (ASL v4i_t*)(void*)l, OFF, 0);
#else
    *(v4u*)((char*)l + OFF) = *(const v4u*)((const char*)g + OFF);
#endif
}

template <int N>
__device__ __forceinline__ void wait_async()
{
#if HAS_ASYNC_LDS
#if __has_builtin(__builtin_amdgcn_s_wait_asynccnt)
    __builtin_amdgcn_s_wait_asynccnt(N);
#else
    if (N == 0) asm volatile("s_wait_asynccnt 0x0" ::: "memory");
    else        asm volatile("s_wait_asynccnt 0x6" ::: "memory");
#endif
#endif
}

__device__ __forceinline__ float gridpt(int j) { return -2.2f + 0.4f * (float)j; }

// ---------------------------------------------------------------------------
// Phase 1a: activation matrix  Act[b, 0:1024] = silu(x), Act[b, 1024 + i*8 + k] = basis
// ---------------------------------------------------------------------------
__global__ __launch_bounds__(256) void kan_prep_act(const float* __restrict__ x,
                                                    bf16_t* __restrict__ act)
{
    const int idx = blockIdx.x * blockDim.x + threadIdx.x;   // 0 .. 8388607
    const int b = idx >> 10;
    const int i = idx & (IN_F - 1);

    const float xv = x[idx];
    const float s  = xv * (1.0f / (1.0f + __expf(-xv)));     // SiLU

    const float xc = fminf(fmaxf(xv, -1.0f), 1.0f);

    // Cox–de Boor, order 3, uniform grid h = 0.4
    float b0[11];
#pragma unroll
    for (int t = 0; t < 11; ++t)
        b0[t] = (xc >= gridpt(t) && xc < gridpt(t + 1)) ? 1.0f : 0.0f;

    float b1[10];
#pragma unroll
    for (int t = 0; t < 10; ++t)
        b1[t] = (xc - gridpt(t)) * (1.0f / 0.4f) * b0[t] +
                (gridpt(t + 2) - xc) * (1.0f / 0.4f) * b0[t + 1];

    float b2[9];
#pragma unroll
    for (int t = 0; t < 9; ++t)
        b2[t] = (xc - gridpt(t)) * (1.0f / 0.8f) * b1[t] +
                (gridpt(t + 3) - xc) * (1.0f / 0.8f) * b1[t + 1];

    float b3[8];
#pragma unroll
    for (int t = 0; t < 8; ++t)
        b3[t] = (xc - gridpt(t)) * (1.0f / 1.2f) * b2[t] +
                (gridpt(t + 4) - xc) * (1.0f / 1.2f) * b2[t + 1];

    bf16_t* row = act + (size_t)b * KTOT;
    row[i] = (bf16_t)s;

    union { v8bf v; bf16_t e[8]; } pk;
#pragma unroll
    for (int k = 0; k < 8; ++k) pk.e[k] = (bf16_t)b3[k];
    *(v8bf*)(row + IN_F + i * NB) = pk.v;                    // 16 B store, aligned
}

// ---------------------------------------------------------------------------
// Phase 1b: packed weight  W[n, 0:1024] = base_w[n,:],  W[n, 1024:] = spline_w[n,:,:]
// ---------------------------------------------------------------------------
__global__ __launch_bounds__(256) void kan_prep_w(const float* __restrict__ bw,
                                                  const float* __restrict__ sw,
                                                  bf16_t* __restrict__ w)
{
    const int n = blockIdx.y;
    const int j = blockIdx.x * blockDim.x + threadIdx.x;     // 0 .. 9215
    const float v = (j < IN_F) ? bw[(size_t)n * IN_F + j]
                               : sw[(size_t)n * (IN_F * NB) + (j - IN_F)];
    w[(size_t)n * KTOT + j] = (bf16_t)v;
}

// ---------------------------------------------------------------------------
// Phase 2: C[8192,1024] = Act[8192,9216] * W[1024,9216]^T  (bf16 WMMA, f32 acc)
// Block tile 128x256, 8 waves each computing 64x64 (4x4 WMMA grid).
// ---------------------------------------------------------------------------
__global__ __launch_bounds__(256) void kan_gemm(const bf16_t* __restrict__ A,
                                                const bf16_t* __restrict__ B,
                                                float* __restrict__ C)
{
    __shared__ bf16_t sA[NBUF][BM * LDSK];   // 3 * 10240 B
    __shared__ bf16_t sB[NBUF][BN * LDSK];   // 3 * 20480 B

    const int tid  = threadIdx.x;
    const int lane = tid & 31;
    const int wave = tid >> 5;
    const int wm   = wave & 1;     // 0..1 : 64-row stripe
    const int wn   = wave >> 1;    // 0..3 : 64-col stripe

    const int mBase = blockIdx.y * BM;
    const int nBase = blockIdx.x * BN;

    // staging assignments:
    //  A tile (128 rows x 64 B): 2 threads/row, 32 B each -> 2 async ops
    //  B tile (256 rows x 64 B): 1 thread/row, 64 B each  -> 4 async ops
    const int lrA = tid >> 1;
    const int lhA = (tid & 1) * 16;      // bf16 elements (32 B granule)

    const bf16_t* gA = A + (size_t)(mBase + lrA) * KTOT + lhA;
    const bf16_t* gB = B + (size_t)(nBase + tid) * KTOT;
    bf16_t* lA = &sA[0][lrA * LDSK + lhA];
    bf16_t* lB = &sB[0][tid * LDSK];
    const int aBuf = BM * LDSK;          // elements per A buffer
    const int bBuf = BN * LDSK;          // elements per B buffer

    v8f acc[4][4];
#pragma unroll
    for (int mt = 0; mt < 4; ++mt)
#pragma unroll
        for (int nt = 0; nt < 4; ++nt)
            acc[mt][nt] = (v8f){0.f, 0.f, 0.f, 0.f, 0.f, 0.f, 0.f, 0.f};

    const int NT = KTOT / BK;            // 288

    // issue one K-tile's DMA group (6 async ops / thread)
    auto issueTile = [&](int kt) {
        const int buf = kt % NBUF;
        const bf16_t* pA = gA + (size_t)kt * BK;
        const bf16_t* pB = gB + (size_t)kt * BK;
        bf16_t* dA = lA + buf * aBuf;
        bf16_t* dB = lB + buf * bBuf;
        cp16<0>(pA, dA);  cp16<16>(pA, dA);
        cp16<0>(pB, dB);  cp16<16>(pB, dB);
        cp16<32>(pB, dB); cp16<48>(pB, dB);
    };

    // prologue: tiles 0 and 1 in flight; wait for tile 0
    issueTile(0);
    issueTile(1);
    wait_async<6>();
    __syncthreads();

    // per-lane fragment addressing (ISA 16-bit A / B layouts, wave32)
    const int frow = lane & 15;
    const int aklo = (lane >> 4) * 8;    // A: K chunks {aklo, aklo+16}
    const int bklo = (lane >> 4) * 8;    // B: K chunks {bklo, bklo+16}

    for (int kt = 0; kt < NT; ++kt) {
        const int cur = kt % NBUF;

        // issue tile kt+2 (its buffer's last readers finished before the
        // previous iteration's barrier)
        if (kt + 2 < NT) issueTile(kt + 2);

        const bf16_t* bufA = &sA[0][0] + cur * aBuf;
        const bf16_t* bufB = &sB[0][0] + cur * bBuf;

        v16bf aFrag[4];
#pragma unroll
        for (int mt = 0; mt < 4; ++mt) {
            union { v16bf v; v8bf h[2]; } u;
            const bf16_t* base = bufA + (wm * 64 + mt * 16 + frow) * LDSK;
            u.h[0] = *(const v8bf*)(base + aklo);
            u.h[1] = *(const v8bf*)(base + 16 + aklo);
            aFrag[mt] = u.v;
        }
#pragma unroll
        for (int nt = 0; nt < 4; ++nt) {
            union { v16bf v; v8bf h[2]; } u;
            const bf16_t* base = bufB + (wn * 64 + nt * 16 + frow) * LDSK;
            u.h[0] = *(const v8bf*)(base + bklo);
            u.h[1] = *(const v8bf*)(base + 16 + bklo);
            const v16bf bFrag = u.v;
#pragma unroll
            for (int mt = 0; mt < 4; ++mt)
                acc[mt][nt] = __builtin_amdgcn_wmma_f32_16x16x32_bf16(
                    false, aFrag[mt], false, bFrag,
                    (short)0, acc[mt][nt], false, false);
        }

        // ensure tile kt+1 is resident (tile kt+2 may stay in flight)
        if (kt + 2 < NT)      wait_async<6>();
        else if (kt + 1 < NT) wait_async<0>();
        __syncthreads();
    }

    // epilogue: C/D layout -> lane 0-15: M=vgpr, lane 16-31: M=8+vgpr; N=lane&15
    const int col  = nBase + wn * 64 + (lane & 15);
    const int rowb = mBase + wm * 64 + ((lane >> 4) * 8);
#pragma unroll
    for (int mt = 0; mt < 4; ++mt)
#pragma unroll
        for (int nt = 0; nt < 4; ++nt)
#pragma unroll
            for (int j = 0; j < 8; ++j)
                C[(size_t)(rowb + mt * 16 + j) * OUT_F + col + nt * 16] = acc[mt][nt][j];
}

// ---------------------------------------------------------------------------
extern "C" void kernel_launch(void* const* d_in, const int* in_sizes, int n_in,
                              void* d_out, int out_size, void* d_ws, size_t ws_size,
                              hipStream_t stream)
{
    const float* x  = (const float*)d_in[0];
    const float* bw = (const float*)d_in[1];
    const float* sw = (const float*)d_in[2];
    float* out = (float*)d_out;

    bf16_t* act = (bf16_t*)d_ws;                                      // 8192*9216*2 B
    bf16_t* w   = (bf16_t*)((char*)d_ws + (size_t)BATCH * KTOT * 2);  // 1024*9216*2 B

    kan_prep_act<<<(BATCH * IN_F) / 256, 256, 0, stream>>>(x, act);
    kan_prep_w<<<dim3(KTOT / 256, OUT_F), 256, 0, stream>>>(bw, sw, w);
    kan_gemm<<<dim3(OUT_F / BN, BATCH / BM), 256, 0, stream>>>(act, w, out);
}
